// DeformNet_60473139528003
// MI455X (gfx1250) — compile-verified, matrix operand order checked by reading, not run on hardware
//
#include <hip/hip_runtime.h>
#include <hip/hip_bf16.h>

// ---------- types for WMMA / async copy ----------
typedef __attribute__((ext_vector_type(16))) _Float16 v16h;
typedef __attribute__((ext_vector_type(8)))  _Float16 v8h;
typedef __attribute__((ext_vector_type(8)))  float    v8f;
typedef __attribute__((ext_vector_type(4)))  int      v4i;

#define NTOT 32768   // 2048 images * 16 spatial positions, N index n = b*16 + p
#define BATCH 2048
#define NT 2         // N-tiles (16 columns each) per block

#if defined(__HIP_DEVICE_COMPILE__)
#if __has_builtin(__builtin_amdgcn_global_load_async_to_lds_b128)
#define HAS_ASYNC_LDS 1
#else
#define HAS_ASYNC_LDS 0
#endif
#else
#define HAS_ASYNC_LDS 0
#endif

// =====================================================================
// WMMA implicit-GEMM:  C[M][N] = A[M][K] * B[K][N]
//   A row-major [Mpad][Kpad] f16 (weights, k = ci*9 + tap), streamed from L2
//   B per-column rows [n][Kpad] f16 (im2col / x_off)
// Each block: MT waves (one 16-row M tile each) x NT 16-col N tiles.
// The 32-row B tile (contiguous in memory) is staged once into LDS
// (async global->LDS when available) and shared by all waves: cuts B
// L2 traffic by MT x, A traffic by NT x.
// Fragment layout (16-bit 16x32 operands):
//   lanes 0-15 : elems 0-7 <- K k0+0..7,   elems 8-15 <- K k0+16..23
//   lanes 16-31: elems 0-7 <- K k0+8..15,  elems 8-15 <- K k0+24..31
// => two contiguous 16B runs per lane: global_load_b128 / ds_load_b128.
// mode 0: out[m*Ntot + n] = relu(acc)          (convs, deform conv)
// mode 1: out[n*18 + m]   = acc + bias[m], m<18 (offset conv)
// =====================================================================
__global__ __launch_bounds__(256)
void gemm_wmma(const _Float16* __restrict__ A, const _Float16* __restrict__ B,
               float* __restrict__ out, const float* __restrict__ bias,
               int Kpad, int mode)
{
    extern __shared__ __align__(16) _Float16 ldsB[];   // [NT*16][Kpad]

    const int lane = threadIdx.x & 31;
    const int wave = threadIdx.x >> 5;      // M tile index
    const int half = lane >> 4;
    const int r    = lane & 15;

    // ---- stage B tile (NT*16 consecutive rows = contiguous block) into LDS ----
    const _Float16* gB = B + (size_t)blockIdx.x * (NT * 16) * Kpad;
    const int chunks = (NT * 16 * Kpad) >> 3;          // 16-byte chunks
    for (int c = threadIdx.x; c < chunks; c += blockDim.x) {
#if HAS_ASYNC_LDS
        __builtin_amdgcn_global_load_async_to_lds_b128(
            (v4i*)(gB + c * 8),
            (__attribute__((address_space(3))) v4i*)(ldsB + c * 8),
            0, 0);
#else
        *(v8h*)(ldsB + c * 8) = *(const v8h*)(gB + c * 8);
#endif
    }
#if HAS_ASYNC_LDS
#if __has_builtin(__builtin_amdgcn_s_wait_asynccnt)
    __builtin_amdgcn_s_wait_asynccnt(0);
#else
    asm volatile("s_wait_asynccnt 0x0" ::: "memory");
#endif
#endif
    __syncthreads();

    const _Float16* arow = A + (size_t)(wave * 16 + r) * Kpad + 8 * half;

    v8f acc[NT];
#pragma unroll
    for (int t = 0; t < NT; ++t) acc[t] = (v8f){};

    for (int k0 = 0; k0 < Kpad; k0 += 32) {
        v8h alo = *(const v8h*)(arow + k0);
        v8h ahi = *(const v8h*)(arow + k0 + 16);
        v16h a;
#pragma unroll
        for (int i = 0; i < 8; ++i) { a[i] = alo[i]; a[i + 8] = ahi[i]; }
#pragma unroll
        for (int t = 0; t < NT; ++t) {
            const _Float16* brow = ldsB + (size_t)(t * 16 + r) * Kpad + 8 * half + k0;
            v8h blo = *(const v8h*)(brow);
            v8h bhi = *(const v8h*)(brow + 16);
            v16h b;
#pragma unroll
            for (int i = 0; i < 8; ++i) { b[i] = blo[i]; b[i + 8] = bhi[i]; }
            acc[t] = __builtin_amdgcn_wmma_f32_16x16x32_f16(
                         false, a, false, b, (short)0, acc[t], false, false);
        }
    }

#pragma unroll
    for (int t = 0; t < NT; ++t) {
        const int n = (blockIdx.x * NT + t) * 16 + r;
        if (mode == 0) {
#pragma unroll
            for (int reg = 0; reg < 8; ++reg) {
                int m = wave * 16 + reg + 8 * half;   // D layout: VGPR reg -> M = reg + 8*half
                float v = acc[t][reg];
                out[(size_t)m * NTOT + n] = v > 0.f ? v : 0.f;
            }
        } else {
#pragma unroll
            for (int reg = 0; reg < 8; ++reg) {
                int m = wave * 16 + reg + 8 * half;
                if (m < 18) out[(size_t)n * 18 + m] = acc[t][reg] + bias[m];
            }
        }
    }
}

// ---------- pack OIHW f32 weights -> [Mpad][Kpad] f16, k = ci*9 + kh*3 + kw ----------
__global__ void pack_w(const float* __restrict__ W, _Float16* __restrict__ A,
                       int Cout, int Cin, int Mpad, int Kpad)
{
    int idx = blockIdx.x * blockDim.x + threadIdx.x;
    if (idx >= Mpad * Kpad) return;
    int k = idx % Kpad, m = idx / Kpad;
    float v = 0.f;
    if (m < Cout && k < Cin * 9) v = W[(size_t)m * Cin * 9 + k];
    A[idx] = (_Float16)v;
}

// ---------- im2col of raw input x (b,3,4,4) -> B[n][32] f16 (K=27 padded) ----------
__global__ void im2col_x(const float* __restrict__ x, _Float16* __restrict__ Bo)
{
    int idx = blockIdx.x * blockDim.x + threadIdx.x;
    if (idx >= NTOT * 32) return;
    int k = idx & 31, n = idx >> 5;
    float val = 0.f;
    if (k < 27) {
        int ci = k / 9, t = k - ci * 9, kh = t / 3, kw = t - kh * 3;
        int p = n & 15, b = n >> 4, oh = p >> 2, ow = p & 3;
        int ih = oh + kh - 1, iw = ow + kw - 1;
        if (ih >= 0 && ih < 4 && iw >= 0 && iw < 4)
            val = x[(size_t)(b * 3 + ci) * 16 + ih * 4 + iw];
    }
    Bo[idx] = (_Float16)val;
}

// ---------- per-channel BN stats (deterministic tree reduction) ----------
__global__ void chan_stats(const float* __restrict__ z, float* __restrict__ stats, int C)
{
    int c = blockIdx.x;
    float s = 0.f, s2 = 0.f;
    for (int i = threadIdx.x; i < NTOT; i += 256) {
        float v = z[(size_t)c * NTOT + i];
        s += v; s2 += v * v;
    }
    __shared__ float sh[256], sh2[256];
    sh[threadIdx.x] = s; sh2[threadIdx.x] = s2;
    __syncthreads();
    for (int st = 128; st > 0; st >>= 1) {
        if (threadIdx.x < st) {
            sh[threadIdx.x]  += sh[threadIdx.x + st];
            sh2[threadIdx.x] += sh2[threadIdx.x + st];
        }
        __syncthreads();
    }
    if (threadIdx.x == 0) {
        float m = sh[0] * (1.f / NTOT);
        float var = sh2[0] * (1.f / NTOT) - m * m;   // biased var (jnp.var)
        stats[c] = m;
        stats[C + c] = rsqrtf(var + 1e-5f);
    }
}

// ---------- fused BN-normalize + im2col pack for next conv ----------
__global__ void bn_im2col(const float* __restrict__ z, const float* __restrict__ stats,
                          const float* __restrict__ g, const float* __restrict__ beta,
                          _Float16* __restrict__ Bo, int Cin, int Kpad)
{
    long idx = (long)blockIdx.x * blockDim.x + threadIdx.x;
    if (idx >= (long)NTOT * Kpad) return;
    int k = (int)(idx % Kpad), n = (int)(idx / Kpad);
    float val = 0.f;
    if (k < Cin * 9) {
        int ci = k / 9, t = k - ci * 9, kh = t / 3, kw = t - kh * 3;
        int p = n & 15, b = n >> 4, oh = p >> 2, ow = p & 3;
        int ih = oh + kh - 1, iw = ow + kw - 1;
        if (ih >= 0 && ih < 4 && iw >= 0 && iw < 4) {
            float v = z[(size_t)ci * NTOT + (b << 4) + ih * 4 + iw];
            val = (v - stats[ci]) * stats[Cin + ci] * g[ci] + beta[ci];
        }
    }
    Bo[idx] = (_Float16)val;
}

__global__ void zero_f32(float* __restrict__ p, int count)
{
    int idx = blockIdx.x * blockDim.x + threadIdx.x;
    if (idx < count) p[idx] = 0.f;
}

// ---------- BN-normalize conv3 into zero-padded 6x6 image (for bilinear sampling) ----------
__global__ void bn_xpad(const float* __restrict__ z, const float* __restrict__ stats,
                        const float* __restrict__ g, const float* __restrict__ beta,
                        float* __restrict__ xpad)
{
    int idx = blockIdx.x * blockDim.x + threadIdx.x;     // (b,ci,p)
    if (idx >= BATCH * 128 * 16) return;
    int p = idx & 15, ci = (idx >> 4) & 127, b = idx >> 11;
    float v = z[(size_t)ci * NTOT + (b << 4) + p];
    float nv = (v - stats[ci]) * stats[128 + ci] * g[ci] + beta[ci];
    int oh = p >> 2, ow = p & 3;
    xpad[(size_t)(b * 128 + ci) * 36 + (oh + 1) * 6 + (ow + 1)] = nv;
}

// ---------- deformable bilinear sampling -> x_off as B operand [n][ci*9+tap] f16 ----------
__global__ void deform_sample(const float* __restrict__ xpad, const float* __restrict__ offs,
                              _Float16* __restrict__ Bo)
{
    int idx = blockIdx.x * blockDim.x + threadIdx.x;     // (n, tap)
    if (idx >= NTOT * 9) return;
    int tap = idx % 9, n = idx / 9;
    int b = n >> 4, p = n & 15, oh = p >> 2, ow = p & 3;

    float offx = offs[(size_t)n * 18 + tap];
    float offy = offs[(size_t)n * 18 + 9 + tap];
    float px = offx + (float)(oh + 1) + (float)(tap / 3 - 1);
    float py = offy + (float)(ow + 1) + (float)(tap % 3 - 1);

    float fx = floorf(px), fy = floorf(py);
    int qltx = min(max((int)fx, 0), 5);
    int qlty = min(max((int)fy, 0), 5);
    int qrbx = min(max((int)(fx + 1.f), 0), 5);
    int qrby = min(max((int)(fy + 1.f), 0), 5);
    float pxc = fminf(fmaxf(px, 0.f), 5.f);
    float pyc = fminf(fmaxf(py, 0.f), 5.f);
    float glt = (1.f + ((float)qltx - pxc)) * (1.f + ((float)qlty - pyc));
    float grb = (1.f - ((float)qrbx - pxc)) * (1.f - ((float)qrby - pyc));
    float glb = (1.f + ((float)qltx - pxc)) * (1.f - ((float)qrby - pyc));
    float grt = (1.f - ((float)qrbx - pxc)) * (1.f + ((float)qlty - pyc));

    const float* xb = xpad + (size_t)b * 128 * 36;
    for (int ci = 0; ci < 128; ++ci) {
        const float* im = xb + ci * 36;
        float v = glt * im[qltx * 6 + qlty] + grb * im[qrbx * 6 + qrby]
                + glb * im[qltx * 6 + qrby] + grt * im[qrbx * 6 + qlty];
        Bo[(size_t)n * 1152 + ci * 9 + tap] = (_Float16)v;
    }
}

// ---------- fused BN + 4x4 average pool ----------
__global__ void bn_pool(const float* __restrict__ z, const float* __restrict__ stats,
                        const float* __restrict__ g, const float* __restrict__ beta,
                        float* __restrict__ pooled)
{
    int idx = blockIdx.x * blockDim.x + threadIdx.x;     // (b, co)
    if (idx >= BATCH * 128) return;
    int co = idx & 127, b = idx >> 7;
    float s = 0.f;
    const float* zr = z + (size_t)co * NTOT + (b << 4);
    for (int p = 0; p < 16; ++p) s += zr[p];
    pooled[idx] = (s * (1.f / 16.f) - stats[co]) * stats[128 + co] * g[co] + beta[co];
}

// ---------- final linear head ----------
__global__ void linear10(const float* __restrict__ pooled, const float* __restrict__ Wc,
                         const float* __restrict__ bc, float* __restrict__ out)
{
    int idx = blockIdx.x * blockDim.x + threadIdx.x;     // (b, k)
    if (idx >= BATCH * 10) return;
    int k = idx % 10, b = idx / 10;
    float s = bc[k];
    const float* pr = pooled + (size_t)b * 128;
    const float* wr = Wc + (size_t)k * 128;
    for (int c = 0; c < 128; ++c) s += pr[c] * wr[c];
    out[idx] = s;
}

// =====================================================================
static inline size_t al256(size_t x) { return (x + 255) & ~(size_t)255; }

extern "C" void kernel_launch(void* const* d_in, const int* in_sizes, int n_in,
                              void* d_out, int out_size, void* d_ws, size_t ws_size,
                              hipStream_t stream)
{
    (void)in_sizes; (void)n_in; (void)out_size; (void)ws_size;
    const float* x  = (const float*)d_in[0];
    const float* W1 = (const float*)d_in[1];
    const float* g1 = (const float*)d_in[2];  const float* b1 = (const float*)d_in[3];
    const float* W2 = (const float*)d_in[4];
    const float* g2 = (const float*)d_in[5];  const float* b2 = (const float*)d_in[6];
    const float* W3 = (const float*)d_in[7];
    const float* g3 = (const float*)d_in[8];  const float* b3 = (const float*)d_in[9];
    const float* Wp = (const float*)d_in[10]; const float* bp = (const float*)d_in[11];
    const float* Wd = (const float*)d_in[12];
    const float* g4 = (const float*)d_in[13]; const float* b4 = (const float*)d_in[14];
    const float* Wc = (const float*)d_in[15]; const float* bc = (const float*)d_in[16];
    float* out = (float*)d_out;

    // ---- workspace carve-up ----
    char* ws = (char*)d_ws;
    size_t off = 0;
    _Float16* A1 = (_Float16*)(ws + off); off += al256((size_t)32 * 32 * 2);
    _Float16* A2 = (_Float16*)(ws + off); off += al256((size_t)64 * 288 * 2);
    _Float16* A3 = (_Float16*)(ws + off); off += al256((size_t)128 * 576 * 2);
    _Float16* Ap = (_Float16*)(ws + off); off += al256((size_t)32 * 1152 * 2);
    _Float16* Ad = (_Float16*)(ws + off); off += al256((size_t)128 * 1152 * 2);
    _Float16* Bb = (_Float16*)(ws + off); off += al256((size_t)NTOT * 1152 * 2);   // 75.5 MB, reused
    float* Z     = (float*)(ws + off);    off += al256((size_t)128 * NTOT * 4);    // 16.8 MB, reused
    float* XP    = (float*)(ws + off);    off += al256((size_t)BATCH * 128 * 36 * 4); // 37.7 MB
    float* OFFS  = (float*)(ws + off);    off += al256((size_t)NTOT * 18 * 4);
    float* ST1   = (float*)(ws + off);    off += al256(256 * 4);
    float* ST2   = (float*)(ws + off);    off += al256(256 * 4);
    float* ST3   = (float*)(ws + off);    off += al256(256 * 4);
    float* ST4   = (float*)(ws + off);    off += al256(256 * 4);
    float* POOL  = (float*)(ws + off);    off += al256((size_t)BATCH * 128 * 4);

    const int T = 256;
    #define GRID1(total) dim3(((total) + T - 1) / T)
    #define LDSB(Kpad) ((size_t)NT * 16 * (Kpad) * 2)   // dynamic LDS bytes for B tile
    const dim3 gGrid(NTOT / (16 * NT));

    // ---- pack all weights ----
    pack_w<<<GRID1(32 * 32),    T, 0, stream>>>(W1, A1, 32, 3, 32, 32);
    pack_w<<<GRID1(64 * 288),   T, 0, stream>>>(W2, A2, 64, 32, 64, 288);
    pack_w<<<GRID1(128 * 576),  T, 0, stream>>>(W3, A3, 128, 64, 128, 576);
    pack_w<<<GRID1(32 * 1152),  T, 0, stream>>>(Wp, Ap, 18, 128, 32, 1152);
    pack_w<<<GRID1(128 * 1152), T, 0, stream>>>(Wd, Ad, 128, 128, 128, 1152);

    // ---- conv1 + relu + BN ----
    im2col_x<<<GRID1(NTOT * 32), T, 0, stream>>>(x, Bb);
    gemm_wmma<<<gGrid, dim3(64),  LDSB(32),  stream>>>(A1, Bb, Z, nullptr, 32, 0);
    chan_stats<<<dim3(32), dim3(256), 0, stream>>>(Z, ST1, 32);
    bn_im2col<<<GRID1(NTOT * 288), T, 0, stream>>>(Z, ST1, g1, b1, Bb, 32, 288);

    // ---- conv2 + relu + BN ----
    gemm_wmma<<<gGrid, dim3(128), LDSB(288), stream>>>(A2, Bb, Z, nullptr, 288, 0);
    chan_stats<<<dim3(64), dim3(256), 0, stream>>>(Z, ST2, 64);
    bn_im2col<<<GRID1(NTOT * 576), T, 0, stream>>>(Z, ST2, g2, b2, Bb, 64, 576);

    // ---- conv3 + relu + BN (-> im2col for offset conv, + padded image for sampling) ----
    gemm_wmma<<<gGrid, dim3(256), LDSB(576), stream>>>(A3, Bb, Z, nullptr, 576, 0);
    chan_stats<<<dim3(128), dim3(256), 0, stream>>>(Z, ST3, 128);
    bn_im2col<<<GRID1(NTOT * 1152), T, 0, stream>>>(Z, ST3, g3, b3, Bb, 128, 1152);
    zero_f32<<<GRID1(BATCH * 128 * 36), T, 0, stream>>>(XP, BATCH * 128 * 36);
    bn_xpad<<<GRID1(BATCH * 128 * 16), T, 0, stream>>>(Z, ST3, g3, b3, XP);

    // ---- offset conv (M=18 padded to 32) ----
    gemm_wmma<<<gGrid, dim3(64),  LDSB(1152), stream>>>(Ap, Bb, OFFS, bp, 1152, 1);

    // ---- deformable bilinear sampling -> B operand for deform GEMM ----
    deform_sample<<<GRID1(NTOT * 9), T, 0, stream>>>(XP, OFFS, Bb);

    // ---- deform conv (stride-ks conv == GEMM over K=1152) + relu + BN + pool ----
    gemm_wmma<<<gGrid, dim3(256), LDSB(1152), stream>>>(Ad, Bb, Z, nullptr, 1152, 0);
    chan_stats<<<dim3(128), dim3(256), 0, stream>>>(Z, ST4, 128);
    bn_pool<<<GRID1(BATCH * 128), T, 0, stream>>>(Z, ST4, g4, b4, POOL);

    // ---- linear head ----
    linear10<<<GRID1(BATCH * 10), T, 0, stream>>>(POOL, Wc, bc, out);
}
